// DTPViT_61014305407184
// MI455X (gfx1250) — matrix operand order repaired and checked by reading, not compiled, not run on hardware
//
#include <hip/hip_runtime.h>

#define DEV __device__ __forceinline__

typedef __attribute__((ext_vector_type(16))) __bf16          v16bf;
typedef __attribute__((ext_vector_type(16))) unsigned short  v16us;
typedef __attribute__((ext_vector_type(8)))  float           v8f;
typedef __attribute__((ext_vector_type(4)))  unsigned int    v4u;
typedef __attribute__((ext_vector_type(8)))  int             v8i;
typedef __attribute__((ext_vector_type(4)))  int             v4i;

#if defined(__has_builtin)
#if __has_builtin(__builtin_amdgcn_tensor_load_to_lds) && __has_builtin(__builtin_amdgcn_s_wait_tensorcnt)
#define HAVE_TDM 1
#endif
#endif
#ifndef HAVE_TDM
#define HAVE_TDM 0
#endif

// ---- model constants ----
#define BB   64
#define LL   196
#define LP   208                 // L padded to 16  (row padding)
#define DD   768
#define HH   12
#define DHH  64
#define FFD  3072
#define NCC  1000
#define NCP  1024                // NCC padded to 128 (TDM tile rows)
#define LJ   224                 // padded attention j / K dim (multiple of 32)
#define ROWS (LL*BB)             // 12544 (l-major, b-minor rows)
#define ROWSP (LP*BB)            // 13312 padded rows for attention operands
#define K3   (3*DD)              // 2304

// ---- scalar helpers ----
DEV unsigned short f2bf(float x) {
  unsigned u = __float_as_uint(x);
  if ((u & 0x7fffffffu) > 0x7f800000u) return (unsigned short)((u >> 16) | 0x40u); // quiet NaN
  u += 0x7fffu + ((u >> 16) & 1u);                                                 // RNE
  return (unsigned short)(u >> 16);
}
DEV float bf2f(unsigned short h) { return __uint_as_float(((unsigned)h) << 16); }
DEV float gelu_f(float x) { return 0.5f * x * (1.0f + erff(x * 0.70710678118f)); }

// ---- unconditional vectorized WMMA fragment loaders (buffers are padded) ----
union Frag { v16us u; v4u q[2]; v16bf b; };

// A 16x32 (MxK): lane halves hold 16 consecutive K: two adjacent b128 loads
DEV v16bf load_a(const unsigned short* X, long ld, int r, int kb) {
  Frag t;
  const v4u* p = reinterpret_cast<const v4u*>(X + (size_t)r * ld + kb);
  t.q[0] = p[0];
  t.q[1] = p[1];
  return t.b;
}
// B 32x16 (KxN) from row-major Wt[N][K]: lane halves hold 16 consecutive K: two adjacent b128 loads
DEV v16bf load_b(const unsigned short* X, long ld, int r, int kb) {
  Frag t;
  const v4u* p = reinterpret_cast<const v4u*>(X + (size_t)r * ld + kb);
  t.q[0] = p[0];
  t.q[1] = p[1];
  return t.b;
}

// ---- TDM: issue one 128-row x 32-col bf16 tile load into LDS (D# per ISA 8.3/8.4) ----
#if HAVE_TDM
DEV void tdm_load_tile(unsigned lds_addr, const unsigned short* gaddr, int K, long ldb) {
  unsigned long long ga = (unsigned long long)(size_t)gaddr;
  v4u g0;
  g0[0] = 1u;                                               // count=1, user mode
  g0[1] = lds_addr;                                         // LDS byte address
  g0[2] = (unsigned)(ga & 0xffffffffu);                     // global_addr[31:0]
  g0[3] = (unsigned)((ga >> 32) & 0x01ffffffu) | (2u << 30);// global_addr[56:32] | type=2
  unsigned td0 = (unsigned)K;                               // tensor_dim0 (row length, elems)
  unsigned td1 = 0x00ffffffu;                               // tensor_dim1: large (buffers padded)
  unsigned long long st0 = (unsigned long long)ldb;         // tensor_dim0_stride (elems)
  v8i g1;
  g1[0] = (int)(1u << 16);                                  // data_size=1 (2 bytes)
  g1[1] = (int)((td0 & 0xffffu) << 16);                     // tensor_dim0[15:0] @ bits63:48
  g1[2] = (int)(((td0 >> 16) & 0xffffu) | ((td1 & 0xffffu) << 16));
  g1[3] = (int)(((td1 >> 16) & 0xffffu) | (32u << 16));     // tile_dim0=32 @ bits127:112
  g1[4] = (int)128u;                                        // tile_dim1=128, tile_dim2=0
  g1[5] = (int)(st0 & 0xffffffffu);                         // stride0[31:0]
  g1[6] = (int)((st0 >> 32) & 0xffffu);                     // stride0[47:32], stride1[15:0]=0
  g1[7] = 0;
  v4i z4; z4[0] = 0; z4[1] = 0; z4[2] = 0; z4[3] = 0;
  v8i z8; z8[0] = 0; z8[1] = 0; z8[2] = 0; z8[3] = 0; z8[4] = 0; z8[5] = 0; z8[6] = 0; z8[7] = 0;
  __builtin_amdgcn_tensor_load_to_lds(g0, g1, z4, z4, z8, 0);
}
#endif

// ---- generic batched bf16 WMMA GEMM, 2x2 tiles per wave (attention path) ----
__global__ void wmma_gemm(const unsigned short* __restrict__ A, long lda, long sA,
                          const unsigned short* __restrict__ Bm, long ldb, long sB, int bmod,
                          const float* __restrict__ bias,
                          const float* __restrict__ resid, long ldr, long sR,
                          float* __restrict__ Cf, long ldc, long sC,
                          unsigned short* __restrict__ Cb, long ldcb, long sCb,
                          int M, int N, int K, int act) {
  int lane = threadIdx.x & 31;
  int wid  = blockIdx.x * (blockDim.x >> 5) + (threadIdx.x >> 5);
  int TM = (M + 15) >> 4, TN = (N + 15) >> 4;
  int TM2 = (TM + 1) >> 1, TN2 = (TN + 1) >> 1;
  if (wid >= TM2 * TN2) return;               // uniform per wave
  int tm0 = (wid / TN2) * 2, tn0 = (wid - (wid / TN2) * TN2) * 2;
  int tm1 = tm0 + 1; if (tm1 > TM - 1) tm1 = TM - 1;
  int tn1 = tn0 + 1; if (tn1 > TN - 1) tn1 = TN - 1;
  int z = blockIdx.z;
  const unsigned short* Az = A  + (size_t)z * sA;
  const unsigned short* Bz = Bm + (size_t)(bmod ? (z % bmod) : z) * sB;

  int lid  = lane & 15;
  int half = lane >> 4;
  int ka = half << 4, kb = half << 4;
  int ra0 = tm0 * 16 + lid, ra1 = tm1 * 16 + lid;
  int rb0 = tn0 * 16 + lid, rb1 = tn1 * 16 + lid;

  v8f c00, c01, c10, c11;
#pragma unroll
  for (int i = 0; i < 8; ++i) { c00[i] = 0.f; c01[i] = 0.f; c10[i] = 0.f; c11[i] = 0.f; }

  for (int k0 = 0; k0 < K; k0 += 32) {
    v16bf a0 = load_a(Az, lda, ra0, k0 + ka);
    v16bf a1 = load_a(Az, lda, ra1, k0 + ka);
    v16bf b0 = load_b(Bz, ldb, rb0, k0 + kb);
    v16bf b1 = load_b(Bz, ldb, rb1, k0 + kb);
    c00 = __builtin_amdgcn_wmma_f32_16x16x32_bf16(false, a0, false, b0, (short)0, c00, false, false);
    c01 = __builtin_amdgcn_wmma_f32_16x16x32_bf16(false, a0, false, b1, (short)0, c01, false, false);
    c10 = __builtin_amdgcn_wmma_f32_16x16x32_bf16(false, a1, false, b0, (short)0, c10, false, false);
    c11 = __builtin_amdgcn_wmma_f32_16x16x32_bf16(false, a1, false, b1, (short)0, c11, false, false);
  }

  auto store_tile = [&](const v8f& acc, int tm, int tn) {
    int n = tn * 16 + lid;
    float bv = (bias != nullptr && n < N) ? bias[n] : 0.0f;
    int mb = tm * 16 + (half << 3);
#pragma unroll
    for (int v = 0; v < 8; ++v) {
      int m = mb + v;
      if (m < M && n < N) {
        float val = acc[v] + bv;
        if (resid) val += resid[(size_t)z * sR + (size_t)m * ldr + n];
        if (act == 1) val = gelu_f(val);
        if (Cf) Cf[(size_t)z * sC  + (size_t)m * ldc  + n] = val;
        if (Cb) Cb[(size_t)z * sCb + (size_t)m * ldcb + n] = f2bf(val);
      }
    }
  };
  store_tile(c00, tm0, tn0);
  if (tn1 != tn0) store_tile(c01, tm0, tn1);
  if (tm1 != tm0) store_tile(c10, tm1, tn0);
  if (tm1 != tm0 && tn1 != tn0) store_tile(c11, tm1, tn1);
}

// ---- weight GEMM with TDM-staged B tiles in LDS (32xM x 128xN per block, 4 waves) ----
// B tile (128 rows x 32 K) DMA'd by the Tensor Data Mover into double-buffered LDS;
// wave0 issues descriptors + waits TENSORcnt; all waves read fragments from LDS.
__global__ void wmma_gemm_tdm(const unsigned short* __restrict__ A, long lda,
                              const unsigned short* __restrict__ Bm, long ldb,
                              const float* __restrict__ bias,
                              const float* __restrict__ resid, long ldr,
                              float* __restrict__ Cf, long ldc,
                              unsigned short* __restrict__ Cb, long ldcb,
                              int M, int N, int K, int act, int GN) {
  __shared__ v4u smem[2][512];                // 2 x 8KB tiles
  int bm = blockIdx.x / GN, bn = blockIdx.x - bm * GN;
  int w = threadIdx.x >> 5, lane = threadIdx.x & 31;
  int lid = lane & 15, half = lane >> 4;
  int TM = (M + 15) >> 4;
  int tm0 = bm * 2, tm1 = tm0 + 1; if (tm1 > TM - 1) tm1 = TM - 1;
  int tn0 = bn * 8 + w * 2, tn1 = tn0 + 1;    // global N tiles (stores guarded by N)
  int ra0 = tm0 * 16 + lid, ra1 = tm1 * 16 + lid;
  int ka = half << 4;
  int rl0 = (w * 2) * 16 + lid, rl1 = (w * 2 + 1) * 16 + lid;  // rows within LDS tile
  const unsigned short* Bbase = Bm + (size_t)bn * 128 * ldb;

  v8f c00, c01, c10, c11;
#pragma unroll
  for (int i = 0; i < 8; ++i) { c00[i] = 0.f; c01[i] = 0.f; c10[i] = 0.f; c11[i] = 0.f; }

  int nk = K >> 5;
#if HAVE_TDM
  bool w0 = (w == 0);
  if (w0) tdm_load_tile((unsigned)(size_t)&smem[0][0], Bbase, K, ldb);
#endif
  for (int i = 0; i < nk; ++i) {
    int k0 = i << 5;
#if HAVE_TDM
    if (w0) {
      if (i + 1 < nk) {
        tdm_load_tile((unsigned)(size_t)&smem[(i + 1) & 1][0], Bbase + (k0 + 32), K, ldb);
        __builtin_amdgcn_s_wait_tensorcnt((short)1);   // oldest (current chunk) done
      } else {
        __builtin_amdgcn_s_wait_tensorcnt((short)0);
      }
    }
    __syncthreads();
    const v4u* sb = &smem[i & 1][0];
    Frag tb0, tb1;
    tb0.q[0] = sb[rl0 * 4 + half * 2]; tb0.q[1] = sb[rl0 * 4 + half * 2 + 1];
    tb1.q[0] = sb[rl1 * 4 + half * 2]; tb1.q[1] = sb[rl1 * 4 + half * 2 + 1];
    v16bf b0 = tb0.b, b1 = tb1.b;
#else
    int kb = half << 4;
    v16bf b0 = load_b(Bm, ldb, tn0 * 16 + lid, k0 + kb);
    v16bf b1 = load_b(Bm, ldb, tn1 * 16 + lid, k0 + kb);
#endif
    v16bf a0 = load_a(A, lda, ra0, k0 + ka);
    v16bf a1 = load_a(A, lda, ra1, k0 + ka);
    c00 = __builtin_amdgcn_wmma_f32_16x16x32_bf16(false, a0, false, b0, (short)0, c00, false, false);
    c01 = __builtin_amdgcn_wmma_f32_16x16x32_bf16(false, a0, false, b1, (short)0, c01, false, false);
    c10 = __builtin_amdgcn_wmma_f32_16x16x32_bf16(false, a1, false, b0, (short)0, c10, false, false);
    c11 = __builtin_amdgcn_wmma_f32_16x16x32_bf16(false, a1, false, b1, (short)0, c11, false, false);
#if HAVE_TDM
    __syncthreads();                          // all waves done reading before next DMA overwrites
#endif
  }

  auto store_tile = [&](const v8f& acc, int tm, int tn) {
    int n = tn * 16 + lid;
    float bv = (bias != nullptr && n < N) ? bias[n] : 0.0f;
    int mb = tm * 16 + (half << 3);
#pragma unroll
    for (int v = 0; v < 8; ++v) {
      int m = mb + v;
      if (m < M && n < N) {
        float val = acc[v] + bv;
        if (resid) val += resid[(size_t)m * ldr + n];
        if (act == 1) val = gelu_f(val);
        if (Cf) Cf[(size_t)m * ldc  + n] = val;
        if (Cb) Cb[(size_t)m * ldcb + n] = f2bf(val);
      }
    }
  };
  store_tile(c00, tm0, tn0);
  store_tile(c01, tm0, tn1);
  if (tm1 != tm0) { store_tile(c10, tm1, tn0); store_tile(c11, tm1, tn1); }
}

// ---- elementwise / small kernels ----
__global__ void cvt_bf(const float* __restrict__ in, unsigned short* __restrict__ out, int n) {
  int t = blockIdx.x * blockDim.x + threadIdx.x;
  if (t < n) out[t] = f2bf(in[t]);
}

__global__ void init_scalars(int* dL, int* dS) { *dL = LL; *dS = 1; }

__global__ void im2col(const float* __restrict__ x, unsigned short* __restrict__ col) {
  int t = blockIdx.x * blockDim.x + threadIdx.x;
  if (t >= ROWS * DD) return;
  int row = t / DD, k = t - row * DD;
  int l = row / BB, b = row - l * BB;
  int c = k >> 8, rem = k & 255, p = rem >> 4, q = rem & 15;
  int hh = l / 14, ww = l - hh * 14;
  col[t] = f2bf(x[(((size_t)b * 3 + c) * 224 + (hh * 16 + p)) * 224 + (ww * 16 + q)]);
}

__global__ void posemb(unsigned short* __restrict__ r, const int* __restrict__ lenp) {
  int t = blockIdx.x * blockDim.x + threadIdx.x;
  if (t >= LP * DD) return;
  int row = t / DD, d = t - row * DD;
  int len = *lenp;
  if (row >= len) { r[t] = 0; return; }
  float pos = (float)(len - 1 - row);
  int j = (d < DD / 2) ? d : d - DD / 2;
  float freq = expf(-logf(10000.0f) * (2.0f * (float)j) / (float)DD);
  float s = pos * freq;
  r[t] = f2bf((d < DD / 2) ? sinf(s) : cosf(s));
}

__global__ void prep_attn(const float* __restrict__ qkv, const float* __restrict__ rwb,
                          const float* __restrict__ rrb, unsigned short* __restrict__ qrw,
                          unsigned short* __restrict__ qrr, unsigned short* __restrict__ kk,
                          unsigned short* __restrict__ vT, const int* __restrict__ bound) {
  int t = blockIdx.x * blockDim.x + threadIdx.x;
  if (t >= ROWS * DD) return;
  int row = t / DD, e = t - row * DD;
  const float* src = qkv + (size_t)row * K3;
  float q = src[e], kv = src[DD + e], vv = src[2 * DD + e];
  qrw[t] = f2bf(q + rwb[e]);
  qrr[t] = f2bf(q + rrb[e]);
  kk[t]  = f2bf(kv);
  int l = row / BB, b = row - l * BB;
  if (l < *bound) {
    int hh = e >> 6, dh = e & 63;
    vT[(((size_t)(b * HH + hh)) * DHH + dh) * LJ + l] = f2bf(vv);
  }
}

// softmax over j with rel_shift(BD) fused via index gather; writes all LP x LJ of prob
__global__ void attn_softmax(const float* __restrict__ AC, const float* __restrict__ BD,
                             unsigned short* __restrict__ prob, const int* __restrict__ bound,
                             float scale) {
  int i = blockIdx.x % LP;
  size_t bh = blockIdx.x / LP;
  const float* ac = AC + bh * (size_t)(LP * LJ) + (size_t)i * LJ;
  const float* bd = BD + bh * (size_t)(LP * LJ);
  unsigned short* pr = prob + bh * (size_t)(LP * LJ) + (size_t)i * LJ;
  int n = *bound;
  int tid = threadIdx.x;
  if (i >= n) { for (int j = tid; j < LJ; j += blockDim.x) pr[j] = 0; return; }
  __shared__ float red[256];
  float lmax = -3.4e38f;
  for (int j = tid; j < n; j += blockDim.x) {
    int idx = n + i * n + j;
    int rr = idx / (n + 1);
    int cc = idx - rr * (n + 1);
    float bdv = (cc == 0) ? 0.0f : bd[(size_t)rr * LJ + (cc - 1)];
    lmax = fmaxf(lmax, (ac[j] + bdv) * scale);
  }
  red[tid] = lmax; __syncthreads();
  for (int st = 128; st > 0; st >>= 1) { if (tid < st) red[tid] = fmaxf(red[tid], red[tid + st]); __syncthreads(); }
  float m = red[0]; __syncthreads();
  float lsum = 0.0f;
  for (int j = tid; j < n; j += blockDim.x) {
    int idx = n + i * n + j;
    int rr = idx / (n + 1);
    int cc = idx - rr * (n + 1);
    float bdv = (cc == 0) ? 0.0f : bd[(size_t)rr * LJ + (cc - 1)];
    lsum += expf((ac[j] + bdv) * scale - m);
  }
  red[tid] = lsum; __syncthreads();
  for (int st = 128; st > 0; st >>= 1) { if (tid < st) red[tid] += red[tid + st]; __syncthreads(); }
  float inv = 1.0f / red[0];
  for (int j = tid; j < LJ; j += blockDim.x) {
    if (j < n) {
      int idx = n + i * n + j;
      int rr = idx / (n + 1);
      int cc = idx - rr * (n + 1);
      float bdv = (cc == 0) ? 0.0f : bd[(size_t)rr * LJ + (cc - 1)];
      pr[j] = f2bf(expf((ac[j] + bdv) * scale - m) * inv);
    } else pr[j] = 0;
  }
}

__global__ void layernorm_k(const float* __restrict__ X, const float* __restrict__ g,
                            const float* __restrict__ bta, float* __restrict__ Yf,
                            unsigned short* __restrict__ Yb, int N) {
  int row = blockIdx.x;
  const float* x = X + (size_t)row * N;
  int tid = threadIdx.x;
  __shared__ float r1[256], r2[256];
  float s1 = 0.f, s2 = 0.f;
  for (int j = tid; j < N; j += blockDim.x) { float v = x[j]; s1 += v; s2 += v * v; }
  r1[tid] = s1; r2[tid] = s2; __syncthreads();
  for (int st = 128; st > 0; st >>= 1) { if (tid < st) { r1[tid] += r1[tid + st]; r2[tid] += r2[tid + st]; } __syncthreads(); }
  float mean = r1[0] / N;
  float var  = r2[0] / N - mean * mean;
  float inv  = rsqrtf(var + 1e-5f);
  for (int j = tid; j < N; j += blockDim.x) {
    float v = (x[j] - mean) * inv * g[j] + bta[j];
    if (Yf) Yf[(size_t)row * N + j] = v;
    if (Yb) Yb[(size_t)row * N + j] = f2bf(v);
  }
}

__global__ void bp_dot(const unsigned short* __restrict__ ff, const float* __restrict__ w2,
                       const float* __restrict__ b2, float* __restrict__ logits) {
  int row = blockIdx.x;                      // row = l*BB + b
  int tid = threadIdx.x;
  __shared__ float red[256];
  float s = 0.f;
  for (int j = tid; j < FFD; j += blockDim.x) s += bf2f(ff[(size_t)row * FFD + j]) * w2[j];
  red[tid] = s; __syncthreads();
  for (int st = 128; st > 0; st >>= 1) { if (tid < st) red[tid] += red[tid + st]; __syncthreads(); }
  if (tid == 0) {
    int l = row / BB, b = row - l * BB;
    logits[b * LL + l] = red[0] + b2[0];
  }
}

__global__ void boundary_scan(const float* __restrict__ logits, int* __restrict__ seg,
                              int* __restrict__ dS) {
  int b = threadIdx.x;
  if (b >= BB) return;
  int cnt = 0;
  for (int l = 0; l < LL; ++l) {
    seg[b * LL + l] = cnt;                   // seg = cumsum - hard
    if (logits[b * LL + l] > 0.0f) cnt++;    // sigmoid(x)>0.5 <=> x>0
  }
  atomicMax(dS, cnt);
}

__global__ void downsample_k(const float* __restrict__ h, const int* __restrict__ seg,
                             const int* __restrict__ dS, float* __restrict__ outp) {
  int t = blockIdx.x * blockDim.x + threadIdx.x;
  if (t >= BB * DD) return;
  int b = t / DD, d = t - b * DD;
  int S = *dS;
  int cur = 0, cnt = 0; float sum = 0.f;
  for (int l = 0; l < LL; ++l) {
    int s = seg[b * LL + l];
    if (s != cur) {
      if (cur < S && cnt > 0) outp[((size_t)cur * BB + b) * DD + d] = sum / (float)cnt;
      cur = s; sum = 0.f; cnt = 0;
    }
    sum += h[((size_t)l * BB + b) * DD + d];
    cnt++;
  }
  if (cur < S && cnt > 0) outp[((size_t)cur * BB + b) * DD + d] = sum / (float)cnt;
}

__global__ void pool_mean(const float* __restrict__ h, const int* __restrict__ dS,
                          unsigned short* __restrict__ pooled) {
  int t = blockIdx.x * blockDim.x + threadIdx.x;
  if (t >= BB * DD) return;
  int b = t / DD, d = t - b * DD;
  int S = *dS; if (S < 1) S = 1;
  float s = 0.f;
  for (int i = 0; i < S; ++i) s += h[((size_t)i * BB + b) * DD + d];
  pooled[(size_t)b * DD + d] = f2bf(s / (float)S);
}

// ---- host side ----
static void gemm(hipStream_t st, const void* A, long lda, long sA,
                 const void* Bm, long ldb, long sB, int bmod,
                 const float* bias, const float* resid, long ldr, long sR,
                 float* Cf, long ldc, long sC,
                 void* Cb, long ldcb, long sCb,
                 int M, int N, int K, int act, int batches) {
  int TM = (M + 15) / 16, TN = (N + 15) / 16;
  int TM2 = (TM + 1) / 2, TN2 = (TN + 1) / 2;
  dim3 grid((TM2 * TN2 + 3) / 4, 1, batches);
  wmma_gemm<<<grid, 128, 0, st>>>((const unsigned short*)A, lda, sA,
                                  (const unsigned short*)Bm, ldb, sB, bmod,
                                  bias, resid, ldr, sR, Cf, ldc, sC,
                                  (unsigned short*)Cb, ldcb, sCb, M, N, K, act);
}

static void gemm_w(hipStream_t st, const void* A, long lda, const void* Bm, long ldb,
                   const float* bias, const float* resid, long ldr,
                   float* Cf, long ldc, void* Cb, long ldcb,
                   int M, int N, int K, int act) {
  int TM = (M + 15) / 16;
  int GM = (TM + 1) / 2;
  int GN = (N + 127) / 128;
  dim3 grid(GM * GN);
  wmma_gemm_tdm<<<grid, 128, 0, st>>>((const unsigned short*)A, lda,
                                      (const unsigned short*)Bm, ldb,
                                      bias, resid, ldr, Cf, ldc,
                                      (unsigned short*)Cb, ldcb, M, N, K, act, GN);
}

extern "C" void kernel_launch(void* const* d_in, const int* in_sizes, int n_in,
                              void* d_out, int out_size, void* d_ws, size_t ws_size,
                              hipStream_t stream) {
  (void)in_sizes; (void)n_in; (void)out_size; (void)ws_size;
  const float* x       = (const float*)d_in[0];
  const float* patch_w = (const float*)d_in[1];
  const float* patch_b = (const float*)d_in[2];
  const float* rwb     = (const float*)d_in[3];
  const float* rrb     = (const float*)d_in[4];
  const float* pre[14]; const float* sht[14];
  for (int i = 0; i < 14; ++i) { pre[i] = (const float*)d_in[5 + i]; sht[i] = (const float*)d_in[19 + i]; }
  const float* bp_w1  = (const float*)d_in[33];
  const float* bp_b1  = (const float*)d_in[34];
  const float* bp_w2  = (const float*)d_in[35];
  const float* bp_b2  = (const float*)d_in[36];
  const float* dng    = (const float*)d_in[37];
  const float* dnb    = (const float*)d_in[38];
  const float* head_w = (const float*)d_in[40];
  const float* head_b = (const float*)d_in[41];
  float* out = (float*)d_out;

  // workspace carving (256B aligned => all GEMM bases 16B aligned)
  size_t off = 0;
  auto alloc = [&](size_t bytes) -> void* {
    void* p = (void*)((char*)d_ws + off);
    off += (bytes + 255) & ~(size_t)255;
    return p;
  };
  const size_t SZ_QKV = (size_t)K3 * DD, SZ_SQ = (size_t)DD * DD, SZ_FF = (size_t)FFD * DD;
  unsigned short* wb_patch = (unsigned short*)alloc(SZ_SQ * 2);
  unsigned short* wb_qkv[2] = { (unsigned short*)alloc(2 * SZ_QKV * 2), (unsigned short*)alloc(8 * SZ_QKV * 2) };
  unsigned short* wb_rw [2] = { (unsigned short*)alloc(2 * SZ_SQ  * 2), (unsigned short*)alloc(8 * SZ_SQ  * 2) };
  unsigned short* wb_ow [2] = { (unsigned short*)alloc(2 * SZ_SQ  * 2), (unsigned short*)alloc(8 * SZ_SQ  * 2) };
  unsigned short* wb_f1 [2] = { (unsigned short*)alloc(2 * SZ_FF  * 2), (unsigned short*)alloc(8 * SZ_FF  * 2) };
  unsigned short* wb_f2 [2] = { (unsigned short*)alloc(2 * SZ_FF  * 2), (unsigned short*)alloc(8 * SZ_FF  * 2) };
  unsigned short* wb_bp1  = (unsigned short*)alloc(SZ_FF * 2);
  unsigned short* wb_head = (unsigned short*)alloc((size_t)NCP * DD * 2);   // padded to 1024 rows

  unsigned short* imcol  = (unsigned short*)alloc((size_t)ROWS * DD * 2);
  float*          h_f    = (float*)         alloc((size_t)ROWS * DD * 4);
  unsigned short* h_b    = (unsigned short*)alloc((size_t)ROWS * DD * 2);
  unsigned short* r_b16  = (unsigned short*)alloc((size_t)LP * DD * 2);     // padded rows
  unsigned short* rk_b16 = (unsigned short*)alloc((size_t)LP * DD * 2);     // padded rows
  float*          qkv_f  = (float*)         alloc((size_t)ROWS * K3 * 4);
  unsigned short* qrw    = (unsigned short*)alloc((size_t)ROWSP * DD * 2);  // padded rows
  unsigned short* qrr    = (unsigned short*)alloc((size_t)ROWSP * DD * 2);
  unsigned short* kk     = (unsigned short*)alloc((size_t)ROWSP * DD * 2);
  unsigned short* vT     = (unsigned short*)alloc((size_t)BB * HH * DHH * LJ * 2);
  float*          AC     = (float*)         alloc((size_t)BB * HH * LP * LJ * 4);
  float*          BD     = (float*)         alloc((size_t)BB * HH * LP * LJ * 4);
  unsigned short* prob   = (unsigned short*)alloc((size_t)BB * HH * LP * LJ * 2);
  float*          attn_f = (float*)         alloc((size_t)ROWS * DD * 4);
  unsigned short* attn_b = (unsigned short*)alloc((size_t)ROWS * DD * 2);
  float*          tmp_f  = (float*)         alloc((size_t)ROWS * DD * 4);
  unsigned short* ff_b16 = (unsigned short*)alloc((size_t)ROWS * FFD * 2);
  float*          logits = (float*)         alloc((size_t)BB * LL * 4);
  int*            seg    = (int*)           alloc((size_t)BB * LL * 4);
  int*            dL     = (int*)           alloc(64);
  int*            dS     = (int*)           alloc(64);
  unsigned short* pooled = (unsigned short*)alloc((size_t)BB * DD * 2);

  auto cvt = [&](const float* src, unsigned short* dst, size_t n) {
    cvt_bf<<<(unsigned)((n + 255) / 256), 256, 0, stream>>>(src, dst, (int)n);
  };

  init_scalars<<<1, 1, 0, stream>>>(dL, dS);
  cvt(patch_w, wb_patch, SZ_SQ);
  cvt(pre[0], wb_qkv[0], 2 * SZ_QKV); cvt(sht[0], wb_qkv[1], 8 * SZ_QKV);
  cvt(pre[2], wb_rw [0], 2 * SZ_SQ ); cvt(sht[2], wb_rw [1], 8 * SZ_SQ );
  cvt(pre[4], wb_ow [0], 2 * SZ_SQ ); cvt(sht[4], wb_ow [1], 8 * SZ_SQ );
  cvt(pre[8], wb_f1 [0], 2 * SZ_FF ); cvt(sht[8], wb_f1 [1], 8 * SZ_FF );
  cvt(pre[10], wb_f2[0], 2 * SZ_FF ); cvt(sht[10], wb_f2[1], 8 * SZ_FF );
  cvt(bp_w1, wb_bp1, SZ_FF);
  cvt(head_w, wb_head, (size_t)NCC * DD);
  // zero padding regions read by unconditional fragment loads / TDM tiles
  (void)hipMemsetAsync(wb_head + (size_t)NCC * DD, 0, (size_t)(NCP - NCC) * DD * 2, stream);
  (void)hipMemsetAsync(qrw + (size_t)ROWS * DD, 0, (size_t)(ROWSP - ROWS) * DD * 2, stream);
  (void)hipMemsetAsync(qrr + (size_t)ROWS * DD, 0, (size_t)(ROWSP - ROWS) * DD * 2, stream);
  (void)hipMemsetAsync(kk  + (size_t)ROWS * DD, 0, (size_t)(ROWSP - ROWS) * DD * 2, stream);

  auto run_layer = [&](int st, int i, const float* P[14], const int* bound) {
    const unsigned short* Wq = wb_qkv[st] + (size_t)i * SZ_QKV;
    const unsigned short* Wr = wb_rw [st] + (size_t)i * SZ_SQ;
    const unsigned short* Wo = wb_ow [st] + (size_t)i * SZ_SQ;
    const unsigned short* W1 = wb_f1 [st] + (size_t)i * SZ_FF;
    const unsigned short* W2 = wb_f2 [st] + (size_t)i * SZ_FF;
    // qkv projection (TDM-staged weights)
    gemm_w(stream, h_b, DD, Wq, DD, P[1] + i * K3, nullptr, 0,
           qkv_f, K3, nullptr, 0, ROWS, K3, DD, 0);
    // rk = r @ r_w.T + r_b   (M=LP so all padded rows are defined)
    gemm_w(stream, r_b16, DD, Wr, DD, P[3] + i * DD, nullptr, 0,
           nullptr, 0, rk_b16, DD, LP, DD, DD, 0);
    prep_attn<<<(ROWS * DD + 255) / 256, 256, 0, stream>>>(qkv_f, rwb, rrb, qrw, qrr, kk, vT, bound);
    // AC[b,h,i,j] = (q+rwb) . k   (batched over z=b*H+h; base offset 64*z)
    gemm(stream, qrw, (long)BB * DD, DHH, kk, (long)BB * DD, DHH, 0, nullptr, nullptr, 0, 0,
         AC, LJ, (long)LP * LJ, nullptr, 0, 0, LL, LL, DHH, 0, BB * HH);
    // BD[b,h,i,j] = (q+rrb) . rk[h]   (B batch offset (z%H)*64)
    gemm(stream, qrr, (long)BB * DD, DHH, rk_b16, DD, DHH, HH, nullptr, nullptr, 0, 0,
         BD, LJ, (long)LP * LJ, nullptr, 0, 0, LL, LL, DHH, 0, BB * HH);
    attn_softmax<<<BB * HH * LP, 256, 0, stream>>>(AC, BD, prob, bound, 0.125f);
    // vec = prob @ v  (B operand = vT rows of d; K padded to LJ with zeros)
    gemm(stream, prob, LJ, (long)LP * LJ, vT, LJ, (long)DHH * LJ, 0, nullptr, nullptr, 0, 0,
         attn_f, (long)BB * DD, DHH, attn_b, (long)BB * DD, DHH, LL, DHH, LJ, 0, BB * HH);
    // o-proj + residual, then LN1
    gemm_w(stream, attn_b, DD, Wo, DD, P[5] + i * DD, h_f, DD,
           tmp_f, DD, nullptr, 0, ROWS, DD, DD, 0);
    layernorm_k<<<ROWS, 256, 0, stream>>>(tmp_f, P[6] + i * DD, P[7] + i * DD, h_f, h_b, DD);
    // FF1 (GELU) -> FF2 + residual -> LN2
    gemm_w(stream, h_b, DD, W1, DD, P[9] + i * FFD, nullptr, 0,
           nullptr, 0, ff_b16, FFD, ROWS, FFD, DD, 1);
    gemm_w(stream, ff_b16, FFD, W2, FFD, P[11] + i * DD, h_f, DD,
           tmp_f, DD, nullptr, 0, ROWS, DD, FFD, 0);
    layernorm_k<<<ROWS, 256, 0, stream>>>(tmp_f, P[12] + i * DD, P[13] + i * DD, h_f, h_b, DD);
  };

  // patch embedding
  im2col<<<(ROWS * DD + 255) / 256, 256, 0, stream>>>(x, imcol);
  gemm_w(stream, imcol, DD, wb_patch, DD, patch_b, nullptr, 0,
         h_f, DD, h_b, DD, ROWS, DD, DD, 0);

  // pre stage (qlen = 196)
  posemb<<<(LP * DD + 255) / 256, 256, 0, stream>>>(r_b16, dL);
  (void)hipMemsetAsync(vT, 0, (size_t)BB * HH * DHH * LJ * 2, stream);
  for (int i = 0; i < 2; ++i) run_layer(0, i, pre, dL);

  // boundary predictor + downsample
  gemm_w(stream, h_b, DD, wb_bp1, DD, bp_b1, nullptr, 0,
         nullptr, 0, ff_b16, FFD, ROWS, FFD, DD, 1);
  bp_dot<<<ROWS, 256, 0, stream>>>(ff_b16, bp_w2, bp_b2, logits);
  boundary_scan<<<1, 64, 0, stream>>>(logits, seg, dS);
  (void)hipMemsetAsync(tmp_f, 0, (size_t)ROWS * DD * 4, stream);
  downsample_k<<<(BB * DD + 255) / 256, 256, 0, stream>>>(h_f, seg, dS, tmp_f);
  layernorm_k<<<ROWS, 256, 0, stream>>>(tmp_f, dng, dnb, h_f, h_b, DD);

  // short stage (qlen = S, padded to 196, masked via dS)
  posemb<<<(LP * DD + 255) / 256, 256, 0, stream>>>(r_b16, dS);
  (void)hipMemsetAsync(vT, 0, (size_t)BB * HH * DHH * LJ * 2, stream);
  for (int i = 0; i < 8; ++i) run_layer(1, i, sht, dS);

  // mean pool over s < S, then classifier head (N=1000 guarded, weights padded to 1024)
  pool_mean<<<(BB * DD + 255) / 256, 256, 0, stream>>>(h_f, dS, pooled);
  gemm_w(stream, pooled, DD, wb_head, DD, head_b, nullptr, 0,
         out, NCC, nullptr, 0, BB, NCC, DD, 0);
}
